// LSTMRNN_49039936586196
// MI455X (gfx1250) — compile-verified, matrix operand order checked by reading, not run on hardware
//
#include <hip/hip_runtime.h>
#include <hip/hip_bf16.h>

typedef __attribute__((ext_vector_type(16))) __bf16 v16bf;
typedef __attribute__((ext_vector_type(8)))  __bf16 v8bf;
typedef __attribute__((ext_vector_type(8)))  float  v8f;

#define B_  64
#define L_  512
#define E_  512
#define H_  1024
#define D_  1536   // E+H
#define NG_ 4096   // 4*H
#define NBLK_ 64   // one block per 16 h-columns

// ws layout (bytes):
//   Wp  @ 0         : 4096*1536*2 = 12,582,912   packed bf16 weights
//   xb  @ 12582912  : 64*512*512*2 = 33,554,432  bf16 copy of x
//   hb  @ 46137344  : 2*64*1024*2  = 262,144     bf16 h ping-pong
//   bar @ 46399488  : 64                          grid barrier (cnt, gen)
#define WS_XB_OFF  12582912ULL
#define WS_HB_OFF  46137344ULL
#define WS_BAR_OFF 46399488ULL

// ---------------------------------------------------------------------------
// One-time packs
// ---------------------------------------------------------------------------
__global__ void lstm_pack_weights(const float* __restrict__ Wf,
                                  const float* __restrict__ Wi,
                                  const float* __restrict__ Wo,
                                  const float* __restrict__ Wu,
                                  unsigned short* __restrict__ Wp) {
    size_t idx = (size_t)blockIdx.x * blockDim.x + threadIdx.x;
    if (idx >= (size_t)NG_ * D_) return;
    int n = (int)(idx / D_);
    int k = (int)(idx % D_);
    int g    = (n >> 4) & 3;
    int j16  = n >> 6;
    int col  = n & 15;
    int jcol = j16 * 16 + col;
    const float* W = (g == 0) ? Wf : (g == 1) ? Wi : (g == 2) ? Wo : Wu;
    __bf16 b = (__bf16)W[(size_t)k * H_ + jcol];
    Wp[idx] = __builtin_bit_cast(unsigned short, b);
}

__global__ void lstm_pack_f32_to_bf16(const float* __restrict__ src,
                                      unsigned short* __restrict__ dst,
                                      size_t n) {
    size_t idx = (size_t)blockIdx.x * blockDim.x + threadIdx.x;
    if (idx >= n) return;
    __bf16 b = (__bf16)src[idx];
    dst[idx] = __builtin_bit_cast(unsigned short, b);
}

__device__ __forceinline__ float sigmoidf_(float x) {
    return 1.0f / (1.0f + __expf(-x));
}

// 16-bit fragment loader (A and B share the lane layout, mirrored M<->N):
// lane l<16 holds index l, K in {0..7,16..23}; lane>=16 holds index l-16,
// K in {8..15,24..31}.  `row` points at the row/column base + K-chunk.
__device__ __forceinline__ v16bf load_frag16(const __bf16* __restrict__ row, int khalf) {
    v8bf lo = *reinterpret_cast<const v8bf*>(row + khalf);
    v8bf hi = *reinterpret_cast<const v8bf*>(row + khalf + 16);
    v16bf f;
#pragma unroll
    for (int i = 0; i < 8; ++i) { f[i] = lo[i]; f[8 + i] = hi[i]; }
    return f;
}

// One GEMM phase over NC K-chunks of 32, software-pipelined one chunk ahead:
// chunk c+1's loads are issued before chunk c's 4 WMMAs so the matrix pipe
// overlaps the L2 fetches.
template <int NC>
__device__ __forceinline__ void gemm_phase(const __bf16* __restrict__ wb,
                                           const __bf16* const* ab,
                                           int khalf, v8f* acc) {
    v16bf bcur = load_frag16(wb, khalf);
    v16bf acur[4];
#pragma unroll
    for (int mt = 0; mt < 4; ++mt) acur[mt] = load_frag16(ab[mt], khalf);

#pragma unroll
    for (int c = 0; c < NC; ++c) {
        v16bf bnxt = bcur;
        v16bf anxt[4];
#pragma unroll
        for (int mt = 0; mt < 4; ++mt) anxt[mt] = acur[mt];
        if (c + 1 < NC) {
            bnxt = load_frag16(wb + (c + 1) * 32, khalf);
#pragma unroll
            for (int mt = 0; mt < 4; ++mt)
                anxt[mt] = load_frag16(ab[mt] + (c + 1) * 32, khalf);
        }
#pragma unroll
        for (int mt = 0; mt < 4; ++mt)
            acc[mt] = __builtin_amdgcn_wmma_f32_16x16x32_bf16(
                false, acur[mt], false, bcur, (short)0, acc[mt], false, false);
        bcur = bnxt;
#pragma unroll
        for (int mt = 0; mt < 4; ++mt) acur[mt] = anxt[mt];
    }
}

// Monotonic-counter grid barrier (no resets -> no reuse races).
__device__ __forceinline__ void grid_barrier(unsigned* cnt, unsigned* gen,
                                             unsigned nblocks, unsigned step) {
    __syncthreads();
    if (threadIdx.x == 0) {
        __threadfence();
        unsigned prev = __hip_atomic_fetch_add(cnt, 1u, __ATOMIC_ACQ_REL,
                                               __HIP_MEMORY_SCOPE_AGENT);
        if (prev == nblocks * (step + 1u) - 1u) {
            __hip_atomic_fetch_add(gen, 1u, __ATOMIC_RELEASE,
                                   __HIP_MEMORY_SCOPE_AGENT);
        } else {
            while (__hip_atomic_load(gen, __ATOMIC_ACQUIRE,
                                     __HIP_MEMORY_SCOPE_AGENT) <= step)
                __builtin_amdgcn_s_sleep(2);
        }
        __threadfence();
    }
    __syncthreads();
}

// ---------------------------------------------------------------------------
// Persistent scan kernel: 64 blocks x 512 threads (16 waves).
// Wave (g, ks): gate g; K-slices x:[ks*128,+128) and h:[512+ks*256,+256).
// Block j16 owns h-cols [16*j16, 16*j16+16).
// ---------------------------------------------------------------------------
__global__ void __launch_bounds__(512)
lstm_scan(const unsigned short* __restrict__ Wp_u16,
          const unsigned short* __restrict__ xb_u16,
          unsigned short* __restrict__ hb_u16,
          const float* __restrict__ c0,
          const float* __restrict__ mask,
          const float* __restrict__ bf, const float* __restrict__ bi,
          const float* __restrict__ bo, const float* __restrict__ bu,
          float* __restrict__ hs, float* __restrict__ cs,
          unsigned* __restrict__ bar) {
    __shared__ float lds[16][64][16];  // [gate*4+ks][row][col], 64 KB

    const int tid   = threadIdx.x;
    const int wave  = tid >> 5;        // 0..15
    const int g     = wave >> 2;       // gate
    const int ks    = wave & 3;        // K-slice
    const int lane  = tid & 31;
    const int ln15  = lane & 15;
    const int khalf = (lane >> 4) * 8;
    const int j16   = blockIdx.x;
    const int jbase = j16 * 16;

    const __bf16* Wp = reinterpret_cast<const __bf16*>(Wp_u16);
    const __bf16* xb = reinterpret_cast<const __bf16*>(xb_u16);
    __bf16*       hb = reinterpret_cast<__bf16*>(hb_u16);
    unsigned* cnt = bar;
    unsigned* gen = bar + 16;

    const int nb = j16 * 64 + g * 16;
    const __bf16* wrow  = Wp + (size_t)(nb + ln15) * D_;  // this lane's B column
    const __bf16* wrowx = wrow + ks * 128;                // x-phase K base
    const __bf16* wrowh = wrow + E_ + ks * 256;           // h-phase K base

    const float* bptr = (g == 0) ? bf : (g == 1) ? bi : (g == 2) ? bo : bu;
    const float bias = (ks == 0) ? bptr[jbase + ln15] : 0.0f;

    // Loop-invariant pieces of the A row bases.
    const __bf16* xbase[4];
    size_t hoff[4];
#pragma unroll
    for (int mt = 0; mt < 4; ++mt) {
        int m = mt * 16 + ln15;
        xbase[mt] = xb + (size_t)m * ((size_t)L_ * E_) + ks * 128;
        hoff[mt]  = (size_t)m * H_ + ks * 256;
    }

    for (int t = 0; t < L_; ++t) {
        const __bf16* hread  = hb + (size_t)(t & 1) * (B_ * H_);
        __bf16*       hwrite = hb + (size_t)((t + 1) & 1) * (B_ * H_);

        const __bf16* abx[4];
        const __bf16* abh[4];
#pragma unroll
        for (int mt = 0; mt < 4; ++mt) {
            abx[mt] = xbase[mt] + (size_t)t * E_;
            abh[mt] = hread + hoff[mt];
        }

        v8f acc[4];
#pragma unroll
        for (int mt = 0; mt < 4; ++mt)
#pragma unroll
            for (int r = 0; r < 8; ++r) acc[mt][r] = bias;

        gemm_phase<4>(wrowx, abx, khalf, acc);  // x part: 4 chunks
        gemm_phase<8>(wrowh, abh, khalf, acc);  // h part: 8 chunks

        // Stage partials: acc[mt][r] is (M = mt*16 + r + khalf, N = ln15).
#pragma unroll
        for (int mt = 0; mt < 4; ++mt)
#pragma unroll
            for (int r = 0; r < 8; ++r)
                lds[wave][mt * 16 + r + khalf][ln15] = acc[mt][r];

        __syncthreads();

        // Fused LSTM cell: 1024 elems over 512 threads.
#pragma unroll
        for (int e = 0; e < 2; ++e) {
            int idx = e * 512 + tid;
            int m   = idx >> 4;
            int col = idx & 15;
            float pf = lds[0][m][col] + lds[1][m][col] + lds[2][m][col] + lds[3][m][col];
            float pi = lds[4][m][col] + lds[5][m][col] + lds[6][m][col] + lds[7][m][col];
            float po = lds[8][m][col] + lds[9][m][col] + lds[10][m][col] + lds[11][m][col];
            float pc = lds[12][m][col] + lds[13][m][col] + lds[14][m][col] + lds[15][m][col];
            float fg = sigmoidf_(pf);
            float ig = sigmoidf_(pi);
            float og = sigmoidf_(po);
            float ch = tanhf(pc);
            float cp = (t == 0)
                ? c0[(size_t)m * H_ + jbase + col]
                : cs[(size_t)m * ((size_t)L_ * H_) + (size_t)(t - 1) * H_ + jbase + col];
            float mv = mask[(size_t)m * L_ + t];
            float nc = (fg * cp + ig * ch) * mv;
            float nh = og * tanhf(fg * cp + ig * ch) * mv;
            size_t o = (size_t)m * ((size_t)L_ * H_) + (size_t)t * H_ + jbase + col;
            hs[o] = nh;
            cs[o] = nc;
            hwrite[(size_t)m * H_ + jbase + col] = (__bf16)nh;
        }

        grid_barrier(cnt, gen, NBLK_, (unsigned)t);
    }
}

extern "C" void kernel_launch(void* const* d_in, const int* in_sizes, int n_in,
                              void* d_out, int out_size, void* d_ws, size_t ws_size,
                              hipStream_t stream) {
    const float* x    = (const float*)d_in[0];
    const float* mask = (const float*)d_in[1];
    const float* Wf_w = (const float*)d_in[2];
    const float* Wf_b = (const float*)d_in[3];
    const float* Wi_w = (const float*)d_in[4];
    const float* Wi_b = (const float*)d_in[5];
    const float* Wo_w = (const float*)d_in[6];
    const float* Wo_b = (const float*)d_in[7];
    const float* U_w  = (const float*)d_in[8];
    const float* U_b  = (const float*)d_in[9];
    const float* h0   = (const float*)d_in[10];
    const float* c0   = (const float*)d_in[11];

    float* hs = (float*)d_out;
    float* cs = hs + (size_t)B_ * L_ * H_;

    char* ws = (char*)d_ws;
    unsigned short* Wp  = (unsigned short*)(ws);
    unsigned short* xb  = (unsigned short*)(ws + WS_XB_OFF);
    unsigned short* hb  = (unsigned short*)(ws + WS_HB_OFF);
    unsigned*       bar = (unsigned*)(ws + WS_BAR_OFF);

    hipMemsetAsync(bar, 0, 64, stream);

    {   // pack weights (bf16, transposed, gate-interleaved)
        size_t total = (size_t)NG_ * D_;
        lstm_pack_weights<<<(int)((total + 255) / 256), 256, 0, stream>>>(
            Wf_w, Wi_w, Wo_w, U_w, Wp);
    }
    {   // pack x -> bf16 (same flat layout)
        size_t total = (size_t)B_ * L_ * E_;
        lstm_pack_f32_to_bf16<<<(int)((total + 255) / 256), 256, 0, stream>>>(
            x, xb, total);
    }
    {   // pack h0 -> bf16 ping buffer 0
        size_t total = (size_t)B_ * H_;
        lstm_pack_f32_to_bf16<<<(int)((total + 255) / 256), 256, 0, stream>>>(
            h0, hb, total);
    }

    lstm_scan<<<NBLK_, 512, 0, stream>>>(Wp, xb, hb, c0, mask,
                                         Wf_b, Wi_b, Wo_b, U_b, hs, cs, bar);
}
